// STCell_33045478375456
// MI455X (gfx1250) — compile-verified, matrix operand order checked by reading, not run on hardware
//
#include <hip/hip_runtime.h>

// ---------------------------------------------------------------------------
// STCell for MI455X (gfx1250, wave32).
// Dense [4000x4000] @ [4000, B*fi] adaptive-mix GEMMs on v_wmma_f32_16x16x32_f16
// (f16 in / f32 accumulate); lm kept as f16 (32MB, L2-resident).
// A-operand staged into LDS via the Tensor Data Mover (tensor_load_to_lds,
// TENSORcnt); B streamed from L2. GAT edge ops via L2 float atomics.
// ---------------------------------------------------------------------------

constexpr int kB  = 4;
constexpr int kN  = 4000;      // nodes per batch
constexpr int kF  = 24;        // F_IN
constexpr int kG  = 16;        // GD
constexpr int kNT = kB * kN;   // 16000
constexpr int kE  = 512000;    // edges per graph
constexpr int kKC = 800;       // K-chunk staged in LDS (multiple of 32, 5 chunks)

typedef __attribute__((ext_vector_type(16))) _Float16 v16h;
typedef __attribute__((ext_vector_type(8)))  _Float16 v8h;
typedef __attribute__((ext_vector_type(8)))  float    v8f;
typedef unsigned int u32x4 __attribute__((ext_vector_type(4)));
typedef int          i32x4 __attribute__((ext_vector_type(4)));
typedef int          i32x8 __attribute__((ext_vector_type(8)));
union V16 { v16h v; v8h h[2]; };

__device__ __forceinline__ float leakyf(float x, float s) {
  return x > 0.f ? x : s * x;
}
// order-preserving float<->uint for atomicMax on signed floats
__device__ __forceinline__ unsigned fenc(float f) {
  unsigned u = __float_as_uint(f);
  return (u & 0x80000000u) ? ~u : (u | 0x80000000u);
}
__device__ __forceinline__ float fdec(unsigned u) {
  unsigned b = (u & 0x80000000u) ? (u & 0x7fffffffu) : ~u;
  return __uint_as_float(b);
}

// ---------------------------------------------------------------------------
// lm = softmax(relu(src @ tgt + I), axis=1), stored as f16. One block per row.
// Row (16KB) cached in LDS so tgt is streamed exactly once.
// ---------------------------------------------------------------------------
__global__ void build_lm_kernel(const float* __restrict__ src,
                                const float* __restrict__ tgt,
                                _Float16* __restrict__ lm16) {
  __shared__ float row[kN];
  __shared__ float red[256];
  __shared__ float srow[12];
  const int u = blockIdx.x;
  const int t = threadIdx.x;
  if (t < 12) srow[t] = src[u * 12 + t];
  __syncthreads();
  float lmax = -3.4e38f;
  for (int v = t; v < kN; v += 256) {
    float acc = 0.f;
#pragma unroll
    for (int k = 0; k < 12; ++k) acc += srow[k] * tgt[k * kN + v];
    if (v == u) acc += 1.f;
    float r = fmaxf(acc, 0.f);
    row[v] = r;
    lmax = fmaxf(lmax, r);
  }
  red[t] = lmax;
  __syncthreads();
  for (int s = 128; s > 0; s >>= 1) {
    if (t < s) red[t] = fmaxf(red[t], red[t + s]);
    __syncthreads();
  }
  const float mx = red[0];
  __syncthreads();
  float lsum = 0.f;
  for (int v = t; v < kN; v += 256) {
    float ex = __expf(row[v] - mx);
    row[v] = ex;
    lsum += ex;
  }
  red[t] = lsum;
  __syncthreads();
  for (int s = 128; s > 0; s >>= 1) {
    if (t < s) red[t] += red[t + s];
    __syncthreads();
  }
  const float inv = 1.f / red[0];
  for (int v = t; v < kN; v += 256)
    lm16[(size_t)u * kN + v] = (_Float16)(row[v] * inv);
}

// ---------------------------------------------------------------------------
// Pack RHS transposed: rhsT[j][v] = h[b*N+v][f], j = b*fi+f  (f16, K-contiguous)
// ---------------------------------------------------------------------------
__global__ void packT_kernel(const float* __restrict__ hin,
                             _Float16* __restrict__ rhsT, int fi) {
  const int total = kB * fi * kN;
  int tid = blockIdx.x * blockDim.x + threadIdx.x;
  if (tid >= total) return;
  int j = tid / kN, v = tid - j * kN;
  int b = j / fi, f = j - b * fi;
  rhsT[tid] = (_Float16)hin[(size_t)(b * kN + v) * fi + f];
}

// ---------------------------------------------------------------------------
// mixed[u][j] = sum_v lm[u][v] * rhsT[j][v]   via v_wmma_f32_16x16x32_f16.
// One block per 16-row u-tile; JT waves, one 16-col j-tile each.
// Wave 0 DMAs the shared A block (16 x 800 f16 per chunk) into LDS with the
// Tensor Data Mover; all waves read A fragments from LDS, B from L2.
// `guard` is always 0 at runtime: the guarded store exists only so the
// compiler cannot prove tileA is never written (the TDM intrinsic writes LDS
// through an integer offset the alias analysis cannot see).
// ---------------------------------------------------------------------------
__global__ void wmma_adp_kernel(const _Float16* __restrict__ lm16,
                                const _Float16* __restrict__ rhsT,
                                float* __restrict__ mixed, int Bfi, int guard) {
  __shared__ _Float16 tileA[16 * kKC];        // 25.6 KB
  if (guard) tileA[guard & 1023] = (_Float16)1;   // never executes (guard==0)
  const int ut   = blockIdx.x;
  const int u0   = ut * 16;
  const int lane = threadIdx.x & 31;
  const int jt   = threadIdx.x >> 5;
  const int m    = lane & 15;                 // A row / B col / D col
  const int hi   = lane >> 4;
  const int hoff = hi ? 8 : 0;
  // 16-bit operand layout: lanes 0-15 hold K 0-7 & 16-23; lanes 16-31 K 8-15 & 24-31
  const _Float16* arow = tileA + (size_t)m * kKC + hoff;
  const _Float16* brow = rhsT + (size_t)(jt * 16 + m) * kN + hoff;
  v8f acc = {};

  for (int c = 0; c < kN / kKC; ++c) {
    __syncthreads();                          // prior chunk fully consumed
    if (threadIdx.x < 32) {
      // TDM 2-D descriptor: data_size=2B, tile 800x16, tensor stride 4000 elems
      unsigned long long ga =
          (unsigned long long)(uintptr_t)(lm16 + (size_t)u0 * kN + c * kKC);
      unsigned ldsOff = (unsigned)__builtin_amdgcn_groupstaticsize();
      u32x4 g0;
      g0.x = 1u;                                            // count=1, user mode
      g0.y = ldsOff;                                        // lds_addr (bytes)
      g0.z = (unsigned)(ga & 0xFFFFFFFFull);                // global_addr[31:0]
      g0.w = (unsigned)((ga >> 32) & 0x1FFFFFFull) | (2u << 30); // [56:32]|type=2
      i32x8 g1;
      g1[0] = (int)(1u << 16);                 // data_size=1 (2B); no pad/iterate
      g1[1] = (int)(((unsigned)kKC & 0xFFFFu) << 16);        // tensor_dim0 lo16
      g1[2] = (int)((16u & 0xFFFFu) << 16);                  // tensor_dim1 lo16
      g1[3] = (int)(((unsigned)kKC & 0xFFFFu) << 16);        // tile_dim0 (=800)
      g1[4] = 16;                                            // tile_dim1=16
      g1[5] = kN;                                            // tensor_dim0_stride
      g1[6] = 0;
      g1[7] = 0;
      i32x4 gz4 = {0, 0, 0, 0};
      i32x8 gz8 = {0, 0, 0, 0, 0, 0, 0, 0};
      // clang-23 (amdgpu-toolchain) 6-arg form: (g0, g1, g2, g3, g?, cpol)
      __builtin_amdgcn_tensor_load_to_lds(g0, g1, gz4, gz4, gz8, 0);
      __builtin_amdgcn_s_wait_tensorcnt(0);
    }
    __syncthreads();                          // LDS tile visible to all waves
    asm volatile("" ::: "memory");            // don't reorder across TDM fill
    const _Float16* bk = brow + c * kKC;
#pragma unroll 5
    for (int k0 = 0; k0 < kKC; k0 += 32) {
      V16 A, Bv;
      A.h[0]  = *(const v8h*)(arow + k0);      // ds_load_b128
      A.h[1]  = *(const v8h*)(arow + k0 + 16);
      Bv.h[0] = *(const v8h*)(bk + k0);        // global_load_b128 (L2)
      Bv.h[1] = *(const v8h*)(bk + k0 + 16);
      acc = __builtin_amdgcn_wmma_f32_16x16x32_f16(false, A.v, false, Bv.v,
                                                   (short)0, acc, false, false);
    }
  }
  const int j0 = jt * 16;
#pragma unroll
  for (int r = 0; r < 8; ++r)   // C/D: lane=N col, VGPR r = M row (hi half: +8)
    mixed[(size_t)(u0 + hi * 8 + r) * Bfi + j0 + m] = acc[r];
}

// s = sigmoid(mixed @ lin_w.T + lin_b), reshaped to [NT,GD]
__global__ void lin_sig_kernel(const float* __restrict__ mixed,
                               const float* __restrict__ w,
                               const float* __restrict__ bias,
                               float* __restrict__ sbuf, int fi) {
  int tid = blockIdx.x * blockDim.x + threadIdx.x;
  if (tid >= kNT * kG) return;
  int node = tid / kG, g = tid - node * kG;
  int b = node / kN, v = node - b * kN;
  const float* mr = mixed + (size_t)v * (kB * fi) + b * fi;
  float acc = bias[g];
  for (int f = 0; f < fi; ++f) acc += mr[f] * w[g * fi + f];
  sbuf[tid] = 1.f / (1.f + __expf(-acc));
}

// ---------------------------------------------------------------------------
// Generic small linear: y = x @ W.T (+bias) (+residual)
// ---------------------------------------------------------------------------
__global__ void linear_kernel(const float* __restrict__ x,
                              const float* __restrict__ W,
                              const float* __restrict__ bias,
                              const float* __restrict__ residual,
                              float* __restrict__ y, int rows, int fin, int fout) {
  int tid = blockIdx.x * blockDim.x + threadIdx.x;
  if (tid >= rows * fout) return;
  int n = tid / fout, o = tid - n * fout;
  float acc = bias ? bias[o] : 0.f;
  const float* xr = x + (size_t)n * fin;
  const float* wr = W + (size_t)o * fin;
  for (int i = 0; i < fin; ++i) acc += xr[i] * wr[i];
  if (residual) acc += residual[tid];
  y[tid] = acc;
}

// ------------------------------ GAT kernels --------------------------------
__global__ void gat_coef_kernel(const float* __restrict__ xp,
                                const float* __restrict__ a_s,
                                const float* __restrict__ a_d,
                                float* __restrict__ asrc, float* __restrict__ adst,
                                int H) {
  int tid = blockIdx.x * blockDim.x + threadIdx.x;
  if (tid >= kNT * H) return;
  int n = tid / H, h = tid - n * H;
  const float* xph = xp + (size_t)n * H * kG + h * kG;
  float a = 0.f, b = 0.f;
#pragma unroll
  for (int c = 0; c < kG; ++c) {
    a += xph[c] * a_s[h * kG + c];
    b += xph[c] * a_d[h * kG + c];
  }
  asrc[tid] = a;
  adst[tid] = b;
}

__global__ void seg_init_kernel(const float* __restrict__ asrc,
                                const float* __restrict__ adst,
                                unsigned* __restrict__ segmax,
                                float* __restrict__ segsum, int H) {
  int tid = blockIdx.x * blockDim.x + threadIdx.x;
  if (tid >= kNT * H) return;
  float e = leakyf(asrc[tid] + adst[tid], 0.2f);   // self-loop edge
  segmax[tid] = fenc(e);
  segsum[tid] = 0.f;
}

__global__ void edge_max_kernel(const int* __restrict__ ei,
                                const float* __restrict__ asrc,
                                const float* __restrict__ adst,
                                unsigned* __restrict__ segmax, int H) {
  int tid = blockIdx.x * blockDim.x + threadIdx.x;
  if (tid >= kE) return;
  int s = ei[tid], d = ei[kE + tid];
  for (int h = 0; h < H; ++h) {
    float e = leakyf(asrc[s * H + h] + adst[d * H + h], 0.2f);
    atomicMax(&segmax[d * H + h], fenc(e));
  }
}

__global__ void edge_expsum_kernel(const int* __restrict__ ei,
                                   const float* __restrict__ asrc,
                                   const float* __restrict__ adst,
                                   const unsigned* __restrict__ segmax,
                                   float* __restrict__ segsum,
                                   float* __restrict__ exbuf, int H) {
  int tid = blockIdx.x * blockDim.x + threadIdx.x;
  if (tid >= kE + kNT) return;
  int s, d;
  if (tid < kE) { s = ei[tid]; d = ei[kE + tid]; }
  else          { s = d = tid - kE; }             // appended self-loops
  for (int h = 0; h < H; ++h) {
    float e  = leakyf(asrc[s * H + h] + adst[d * H + h], 0.2f);
    float ex = __expf(e - fdec(segmax[d * H + h]));
    exbuf[(size_t)tid * H + h] = ex;
    atomicAdd(&segsum[d * H + h], ex);
  }
}

__global__ void gat_out_init_kernel(float* __restrict__ gout,
                                    const float* __restrict__ gb) {
  int tid = blockIdx.x * blockDim.x + threadIdx.x;
  if (tid >= kNT * kG) return;
  gout[tid] = gb[tid % kG];                       // out = mean_h(agg) + bias
}

__global__ void edge_agg_kernel(const int* __restrict__ ei,
                                const float* __restrict__ xp,
                                const float* __restrict__ exbuf,
                                const float* __restrict__ segsum,
                                float* __restrict__ gout, int H) {
  int tid = blockIdx.x * blockDim.x + threadIdx.x;
  if (tid >= kE + kNT) return;
  int s, d;
  if (tid < kE) { s = ei[tid]; d = ei[kE + tid]; }
  else          { s = d = tid - kE; }
  const float invH = 1.f / (float)H;
  for (int h = 0; h < H; ++h) {
    float attn = exbuf[(size_t)tid * H + h] / segsum[d * H + h] * invH;
    const float* xs = xp + (size_t)s * H * kG + h * kG;
#pragma unroll
    for (int c = 0; c < kG; ++c)
      atomicAdd(&gout[d * kG + c], xs[c] * attn);
  }
}

// ------------------------- gating / elementwise ----------------------------
// mode: 0=tanh, 1=leaky(0.01), 2=relu
__global__ void combine_kernel(const float* __restrict__ g,
                               const float* __restrict__ s,
                               const float* __restrict__ other,
                               float* __restrict__ o, int mode) {
  int tid = blockIdx.x * blockDim.x + threadIdx.x;
  if (tid >= kNT * kG) return;
  float gv = g[tid];
  float a = (mode == 0) ? tanhf(gv)
          : (mode == 1) ? leakyf(gv, 0.01f)
                        : fmaxf(gv, 0.f);
  float sv = s[tid];
  o[tid] = a * sv + other[tid] * (1.f - sv);
}

__global__ void act_kernel(const float* __restrict__ in, float* __restrict__ out,
                           int mode) {
  int tid = blockIdx.x * blockDim.x + threadIdx.x;
  if (tid >= kNT * kG) return;
  float v = in[tid];
  out[tid] = (mode == 0) ? tanhf(v) : fmaxf(v, 0.f);
}

__global__ void final_kernel(const float* __restrict__ o1, const float* __restrict__ o2,
                             const float* __restrict__ o3, const float* __restrict__ o4,
                             const float* __restrict__ fw, const float* __restrict__ fb,
                             float* __restrict__ cell, int coloff) {
  int tid = blockIdx.x * blockDim.x + threadIdx.x;
  if (tid >= kNT * kG) return;
  int n = tid / kG, g = tid - n * kG;
  const float* w  = fw + g * 64;
  const float* p1 = o1 + (size_t)n * kG;
  const float* p2 = o2 + (size_t)n * kG;
  const float* p3 = o3 + (size_t)n * kG;
  const float* p4 = o4 + (size_t)n * kG;
  float acc = fb[g];
#pragma unroll
  for (int c = 0; c < kG; ++c)
    acc += p1[c] * w[c] + p2[c] * w[16 + c] + p3[c] * w[32 + c] + p4[c] * w[48 + c];
  cell[(size_t)n * 32 + coloff + g] = acc;
}

__global__ void out_kernel(const float* __restrict__ cell,
                           const float* __restrict__ ow,
                           const float* __restrict__ ob,
                           float* __restrict__ out) {
  int tid = blockIdx.x * blockDim.x + threadIdx.x;
  if (tid >= kNT * kF) return;
  int n = tid / kF, o = tid - n * kF;
  float acc = ob[o];
#pragma unroll
  for (int i = 0; i < 32; ++i) acc += cell[(size_t)n * 32 + i] * ow[o * 32 + i];
  out[tid] = acc;
}

// ---------------------------------------------------------------------------
// Host orchestration
// ---------------------------------------------------------------------------
extern "C" void kernel_launch(void* const* d_in, const int* in_sizes, int n_in,
                              void* d_out, int out_size, void* d_ws, size_t ws_size,
                              hipStream_t stream) {
  (void)in_sizes; (void)n_in; (void)out_size; (void)ws_size;
  auto cdiv = [](long a, long b) { return (int)((a + b - 1) / b); };

  // ---- workspace carve (256B aligned slices) ----
  char* wsb = (char*)d_ws;
  size_t off = 0;
  auto carve = [&](size_t bytes) -> void* {
    void* p = wsb + off;
    off = (off + bytes + 255) & ~(size_t)255;
    return p;
  };
  _Float16* lm16  = (_Float16*)carve((size_t)kN * kN * sizeof(_Float16));   // 32MB
  _Float16* rhsT  = (_Float16*)carve((size_t)kB * kF * kN * sizeof(_Float16));
  float* mixed    = (float*)carve((size_t)kN * kB * kF * sizeof(float));
  float* x1       = (float*)carve((size_t)kNT * kF * sizeof(float));
  float* x2       = (float*)carve((size_t)kNT * kF * sizeof(float));
  float* origin   = (float*)carve((size_t)kNT * kG * sizeof(float));
  float* obuf[4];
  for (int i = 0; i < 4; ++i) obuf[i] = (float*)carve((size_t)kNT * kG * sizeof(float));
  float* sbuf     = (float*)carve((size_t)kNT * kG * sizeof(float));
  float* gout     = (float*)carve((size_t)kNT * kG * sizeof(float));
  float* hbuf     = (float*)carve((size_t)kNT * kF * sizeof(float));
  float* xp       = (float*)carve((size_t)kNT * 3 * kG * sizeof(float));
  float* asrc     = (float*)carve((size_t)kNT * 3 * sizeof(float));
  float* adst     = (float*)carve((size_t)kNT * 3 * sizeof(float));
  unsigned* segmx = (unsigned*)carve((size_t)kNT * 3 * sizeof(unsigned));
  float* segsum   = (float*)carve((size_t)kNT * 3 * sizeof(float));
  float* exbuf    = (float*)carve((size_t)(kE + kNT) * 3 * sizeof(float));
  float* cell     = (float*)carve((size_t)kNT * 32 * sizeof(float));

  // ---- inputs (setup_inputs dict order; params flattened in insertion order) ----
  const float* x     = (const float*)d_in[0];
  const int* ei_sp   = (const int*)d_in[1];
  const int* ei_dtw  = (const int*)d_in[2];
  auto pf = [&](int i) { return (const float*)d_in[i]; };
  // 4:seq_w 5:seq_b 6:out_w 7:out_b | 8..35 sp block | 36..63 dtw block | 64:final_w 65:final_b

  // x1 = x @ seq_w.T + seq_b + x ; x2 = x1 @ seq_w.T + seq_b + x1
  linear_kernel<<<cdiv((long)kNT * kF, 256), 256, 0, stream>>>(
      x, pf(4), pf(5), x, x1, kNT, kF, kF);
  linear_kernel<<<cdiv((long)kNT * kF, 256), 256, 0, stream>>>(
      x1, pf(4), pf(5), x1, x2, kNT, kF, kF);

  for (int br = 0; br < 2; ++br) {
    const int po = 8 + br * 28;                 // branch param base
    const float* xin = br ? x2 : x1;
    const int* ei    = br ? ei_dtw : ei_sp;
    const int heads[4] = {3, 3, 3, br ? 3 : 1};
    const int fins[4]  = {kF, kG, kG, kG};

    build_lm_kernel<<<kN, 256, 0, stream>>>(pf(po + 0), pf(po + 1), lm16);
    linear_kernel<<<cdiv((long)kNT * kG, 256), 256, 0, stream>>>(
        xin, pf(po + 2), pf(po + 3), nullptr, origin, kNT, kF, kG);

    const float* hin = xin;
    for (int st = 0; st < 4; ++st) {
      const int fi = fins[st], H = heads[st];
      const int pg = po + 4 + st * 6;           // gW, gas, gad, gb, linw, linb
      const int Bfi = kB * fi;

      // ---- adaptive mix:  s = sigmoid( (lm @ h) @ lin_w.T + lin_b ) ----
      packT_kernel<<<cdiv((long)Bfi * kN, 256), 256, 0, stream>>>(hin, rhsT, fi);
      wmma_adp_kernel<<<kN / 16, 32 * (Bfi / 16), 0, stream>>>(
          lm16, rhsT, mixed, Bfi, 0);
      lin_sig_kernel<<<cdiv((long)kNT * kG, 256), 256, 0, stream>>>(
          mixed, pf(pg + 4), pf(pg + 5), sbuf, fi);

      // ---- GAT (concat=False mean over heads, self-loops, slope 0.2) ----
      linear_kernel<<<cdiv((long)kNT * H * kG, 256), 256, 0, stream>>>(
          hin, pf(pg + 0), nullptr, nullptr, xp, kNT, fi, H * kG);
      gat_coef_kernel<<<cdiv((long)kNT * H, 256), 256, 0, stream>>>(
          xp, pf(pg + 1), pf(pg + 2), asrc, adst, H);
      seg_init_kernel<<<cdiv((long)kNT * H, 256), 256, 0, stream>>>(
          asrc, adst, segmx, segsum, H);
      edge_max_kernel<<<cdiv((long)kE, 256), 256, 0, stream>>>(
          ei, asrc, adst, segmx, H);
      edge_expsum_kernel<<<cdiv((long)(kE + kNT), 256), 256, 0, stream>>>(
          ei, asrc, adst, segmx, segsum, exbuf, H);
      gat_out_init_kernel<<<cdiv((long)kNT * kG, 256), 256, 0, stream>>>(
          gout, pf(pg + 3));
      edge_agg_kernel<<<cdiv((long)(kE + kNT), 256), 256, 0, stream>>>(
          ei, xp, exbuf, segsum, gout, H);

      // ---- gate + residual ----
      int mode = (st == 0) ? 0 : (st == 1) ? 1 : 2;
      const float* other = (st == 0) ? origin : obuf[st - 1];
      combine_kernel<<<cdiv((long)kNT * kG, 256), 256, 0, stream>>>(
          gout, sbuf, other, obuf[st], mode);
      if (st < 3) {
        int amode = (st == 0) ? 0 : 2;          // h = tanh(o1) / relu(o2) / relu(o3)
        act_kernel<<<cdiv((long)kNT * kG, 256), 256, 0, stream>>>(
            obuf[st], hbuf, amode);
        hin = hbuf;
      }
    }
    // shared final linear (reference bug kept): cell cols [br*16, br*16+16)
    final_kernel<<<cdiv((long)kNT * kG, 256), 256, 0, stream>>>(
        obuf[0], obuf[1], obuf[2], obuf[3], pf(64), pf(65), cell, br * 16);
  }

  out_kernel<<<cdiv((long)kNT * kF, 256), 256, 0, stream>>>(
      cell, pf(6), pf(7), (float*)d_out);
}
